// GraphTransformer_43465069035643
// MI455X (gfx1250) — matrix operand order repaired and checked by reading, not compiled, not run on hardware
//
#include <hip/hip_runtime.h>

#define D 128
#define H 8
#define CH 16
#define NLAYERS 4
#define NODE_IN 16
#define NEG_SLOPE 0.2f
#define LN_EPS 1e-5f

typedef __attribute__((ext_vector_type(16))) _Float16 v16h;
typedef __attribute__((ext_vector_type(8)))  float    v8f;
typedef __attribute__((ext_vector_type(4)))  float    v4f;

// order-preserving float <-> uint mapping for atomicMax-based segment_max
__device__ __forceinline__ unsigned fenc(float f) {
  unsigned u = __float_as_uint(f);
  return (u & 0x80000000u) ? ~u : (u | 0x80000000u);
}
__device__ __forceinline__ float fdec(unsigned u) {
  unsigned b = (u & 0x80000000u) ? (u ^ 0x80000000u) : ~u;
  return __uint_as_float(b);
}

__global__ void k_zero(unsigned* __restrict__ p, int n) {
  int i = blockIdx.x * blockDim.x + threadIdx.x;
  if (i < n) p[i] = 0u;
}

// h[n][d] = x[n][:16] @ node_w[:,d] + node_b[d]
__global__ void k_node_embed(const float* __restrict__ x, const float* __restrict__ w,
                             const float* __restrict__ b, float* __restrict__ h, int n) {
  int i = blockIdx.x * blockDim.x + threadIdx.x;
  if (i >= n * D) return;
  int node = i >> 7, d = i & (D - 1);
  float acc = b[d];
#pragma unroll
  for (int k = 0; k < NODE_IN; ++k) acc += x[node * NODE_IN + k] * w[k * D + d];
  h[i] = acc;
}

// wt[m][j][k] = (f16) W_m[k][j]   (m: 0-3 lin, 4-7 ff1, 8-11 ff2) -> K-contiguous per column
__global__ void k_wconv(const float* __restrict__ lin, const float* __restrict__ ff1,
                        const float* __restrict__ ff2, _Float16* __restrict__ wt) {
  int i = blockIdx.x * blockDim.x + threadIdx.x;
  if (i >= 12 * D * D) return;
  int m = i >> 14;
  int e = i & 16383;
  int j = e >> 7, k = e & 127;
  const float* src = (m < 4) ? (lin + m * D * D)
                   : (m < 8) ? (ff1 + (m - 4) * D * D)
                             : (ff2 + (m - 8) * D * D);
  wt[i] = (_Float16)src[k * D + j];
}

// C[N,128] = A[N,128] @ W[128,128] via v_wmma_f32_16x16x32_f16.
// mode: 0 = plain, 1 = +bias then ReLU, 2 = +bias
__global__ __launch_bounds__(256)
void k_gemm128(const float* __restrict__ A, const _Float16* __restrict__ Wt,
               float* __restrict__ Cm, const float* __restrict__ bias,
               int mode, int nrows) {
  int wave = threadIdx.x >> 5;
  int lane = threadIdx.x & 31;
  int r0 = blockIdx.x * 16;
  int c0 = wave * 16;          // 8 waves cover the 128 output columns
  int half = lane >> 4;        // lane half selects K sub-chunk
  int mrow = lane & 15;
  int col = c0 + mrow;

  int arow_i = r0 + mrow;
  if (arow_i >= nrows) arow_i = nrows - 1;   // clamp (N % 16 == 0 normally)
  const float*    arow = A + (long)arow_i * D;
  const _Float16* bcol = Wt + (long)col * D;

  v8f acc = {};
#pragma unroll
  for (int kb = 0; kb < 4; ++kb) {
    // ---- A fragment (16 consecutive K per lane-half, matching B)
    int ka = kb * 32 + half * 16;
    v4f a0 = *(const v4f*)(arow + ka);
    v4f a1 = *(const v4f*)(arow + ka + 4);
    v4f a2 = *(const v4f*)(arow + ka + 8);
    v4f a3 = *(const v4f*)(arow + ka + 12);
    v16h af;
#pragma unroll
    for (int t = 0; t < 4; ++t) {
      af[t]      = (_Float16)a0[t];
      af[4 + t]  = (_Float16)a1[t];
      af[8 + t]  = (_Float16)a2[t];
      af[12 + t] = (_Float16)a3[t];
    }
    // ---- B fragment (K order chosen to match A: 16 consecutive K per lane-half, col = lane&15)
    v16h bf = *(const v16h*)(bcol + kb * 32 + half * 16);

    acc = __builtin_amdgcn_wmma_f32_16x16x32_f16(false, af, false, bf,
                                                 (short)0, acc, false, false);
  }

  float bv = (mode != 0) ? bias[col] : 0.0f;
#pragma unroll
  for (int v = 0; v < 8; ++v) {
    float o = acc[v] + bv;
    if (mode == 1) o = o > 0.0f ? o : 0.0f;
    int orow = r0 + half * 8 + v;
    if (orow < nrows) Cm[(long)orow * D + col] = o;
  }
}

// a_s[n][h], a_d[n][h] = <xh[n,h,:], att_src/att_dst[h,:]>
__global__ void k_attn_coef(const float* __restrict__ xh, const float* __restrict__ asrc,
                            const float* __restrict__ adst, float* __restrict__ a_s,
                            float* __restrict__ a_d, int n) {
  int i = blockIdx.x * blockDim.x + threadIdx.x;
  if (i >= n * H) return;
  int node = i >> 3, hh = i & 7;
  const float* xp = xh + (long)node * D + hh * CH;
  float s = 0.f, d = 0.f;
#pragma unroll
  for (int c = 0; c < CH; ++c) {
    float v = xp[c];
    s += v * asrc[hh * CH + c];
    d += v * adst[hh * CH + c];
  }
  a_s[i] = s;
  a_d[i] = d;
}

// alpha = leaky_relu(a_s[src]+a_d[dst]); segment max into menc (encoded uint)
__global__ void k_edge_alpha(const int* __restrict__ ei, const float* __restrict__ a_s,
                             const float* __restrict__ a_d, float* __restrict__ alpha,
                             unsigned* __restrict__ menc, int E, int n) {
  int i = blockIdx.x * blockDim.x + threadIdx.x;
  if (i >= (E + n) * H) return;
  int e = i >> 3, hh = i & 7;
  int s, d;
  if (e < E) { s = ei[e]; d = ei[E + e]; } else { s = d = e - E; }
  float a = a_s[s * H + hh] + a_d[d * H + hh];
  a = a > 0.f ? a : NEG_SLOPE * a;
  alpha[i] = a;
  atomicMax(&menc[d * H + hh], fenc(a));
}

// e = exp(alpha - m[dst]); segment sum into denom; alpha overwritten with e
__global__ void k_edge_exp(const int* __restrict__ ei, float* __restrict__ alpha,
                           const unsigned* __restrict__ menc, float* __restrict__ denom,
                           int E, int n) {
  int i = blockIdx.x * blockDim.x + threadIdx.x;
  if (i >= (E + n) * H) return;
  int e = i >> 3, hh = i & 7;
  int d = (e < E) ? ei[E + e] : (e - E);
  float m = fdec(menc[d * H + hh]);
  float ew = __expf(alpha[i] - m);
  alpha[i] = ew;
  atomicAdd(&denom[d * H + hh], ew);
}

// agg[dst] += xh[src] * (e / denom[dst]); one wave per edge, 4 channels/lane
__global__ void k_edge_agg(const int* __restrict__ ei, const float* __restrict__ xh,
                           const float* __restrict__ alpha, const float* __restrict__ denom,
                           float* __restrict__ agg, int E, int n) {
  int tid = blockIdx.x * blockDim.x + threadIdx.x;
  int e = tid >> 5;
  if (e >= E + n) return;
  int lane = tid & 31;
  int s, d;
  if (e < E) { s = ei[e]; d = ei[E + e]; } else { s = d = e - E; }
  int c0 = lane * 4;
  int hh = lane >> 2;                       // (lane*4)/16
  float w = alpha[(long)e * H + hh] / denom[d * H + hh];
  v4f xv = *(const v4f*)(xh + (long)s * D + c0);
  float* ap = agg + (long)d * D + c0;
  atomicAdd(ap + 0, xv[0] * w);
  atomicAdd(ap + 1, xv[1] * w);
  atomicAdd(ap + 2, xv[2] * w);
  atomicAdd(ap + 3, xv[3] * w);
}

// h2 = LayerNorm(agg + att_b + h) * g + b    (block = 128 threads = one node)
__global__ void k_ln(const float* __restrict__ agg, const float* __restrict__ attb,
                     const float* __restrict__ hin, const float* __restrict__ g,
                     const float* __restrict__ b, float* __restrict__ hout) {
  int node = blockIdx.x, t = threadIdx.x;
  float v = agg[(long)node * D + t] + attb[t] + hin[(long)node * D + t];
  float s = v, q = v * v;
#pragma unroll
  for (int off = 16; off > 0; off >>= 1) {
    s += __shfl_down(s, off);
    q += __shfl_down(q, off);
  }
  __shared__ float ss[4], sq[4];
  int wid = t >> 5, lane = t & 31;
  if (lane == 0) { ss[wid] = s; sq[wid] = q; }
  __syncthreads();
  float tot  = ss[0] + ss[1] + ss[2] + ss[3];
  float totq = sq[0] + sq[1] + sq[2] + sq[3];
  float mu = tot * (1.0f / D);
  float var = totq * (1.0f / D) - mu * mu;
  float r = rsqrtf(var + LN_EPS);
  hout[(long)node * D + t] = (v - mu) * r * g[t] + b[t];
}

// deterministic two-stage mean over nodes
__global__ void k_mean1(const float* __restrict__ h, float* __restrict__ partial,
                        int n, int per) {
  int b = blockIdx.x, t = threadIdx.x;
  long start = (long)b * per;
  long end = start + per; if (end > n) end = n;
  float s = 0.f;
  for (long node = start; node < end; ++node) s += h[node * D + t];
  partial[b * D + t] = s;
}
__global__ void k_mean2(const float* __restrict__ partial, float* __restrict__ out,
                        int nb, float inv) {
  int t = threadIdx.x;
  float s = 0.f;
  for (int b = 0; b < nb; ++b) s += partial[b * D + t];
  out[t] = s * inv;
}

extern "C" void kernel_launch(void* const* d_in, const int* in_sizes, int n_in,
                              void* d_out, int out_size, void* d_ws, size_t ws_size,
                              hipStream_t stream) {
  const float* x       = (const float*)d_in[0];
  const int*   ei      = (const int*)  d_in[2];
  const float* node_w  = (const float*)d_in[3];
  const float* node_b  = (const float*)d_in[4];
  const float* lin_w   = (const float*)d_in[7];
  const float* att_src = (const float*)d_in[8];
  const float* att_dst = (const float*)d_in[9];
  const float* att_b   = (const float*)d_in[10];
  const float* ln_g    = (const float*)d_in[11];
  const float* ln_b    = (const float*)d_in[12];
  const float* ff1_w   = (const float*)d_in[13];
  const float* ff1_b   = (const float*)d_in[14];
  const float* ff2_w   = (const float*)d_in[15];
  const float* ff2_b   = (const float*)d_in[16];

  int N = in_sizes[0] / NODE_IN;
  int E = in_sizes[2] / 2;

  char* p = (char*)d_ws;
  auto carve = [&](size_t bytes) -> char* {
    char* r = p;
    p += (bytes + 255) & ~(size_t)255;
    return r;
  };
  float*    h     = (float*)   carve((size_t)N * D * 4);
  float*    xh    = (float*)   carve((size_t)N * D * 4);
  float*    h2    = (float*)   carve((size_t)N * D * 4);
  float*    agg   = (float*)   carve((size_t)N * D * 4);
  float*    a_s   = (float*)   carve((size_t)N * H * 4);
  float*    a_d   = (float*)   carve((size_t)N * H * 4);
  unsigned* menc  = (unsigned*)carve((size_t)N * H * 4);
  float*    denom = (float*)   carve((size_t)N * H * 4);
  float*    alpha = (float*)   carve((size_t)(E + N) * H * 4);
  _Float16* wt    = (_Float16*)carve((size_t)12 * D * D * 2);
  float*    part  = (float*)   carve((size_t)128 * D * 4);

  const int T = 256;
  int nd  = N * D;
  int nh  = N * H;
  int teh = (E + N) * H;
  int tew = (E + N) * 32;

  k_wconv<<<(12 * D * D + T - 1) / T, T, 0, stream>>>(lin_w, ff1_w, ff2_w, wt);
  k_node_embed<<<(nd + T - 1) / T, T, 0, stream>>>(x, node_w, node_b, h, N);

  int rowTiles = (N + 15) / 16;
  for (int l = 0; l < NLAYERS; ++l) {
    const _Float16* wlin = wt + (size_t)l * D * D;
    const _Float16* wf1  = wt + (size_t)(4 + l) * D * D;
    const _Float16* wf2  = wt + (size_t)(8 + l) * D * D;

    k_gemm128<<<rowTiles, 256, 0, stream>>>(h, wlin, xh, nullptr, 0, N);
    k_attn_coef<<<(nh + T - 1) / T, T, 0, stream>>>(xh, att_src + l * D, att_dst + l * D,
                                                    a_s, a_d, N);
    k_zero<<<(nd + T - 1) / T, T, 0, stream>>>((unsigned*)agg, nd);
    k_zero<<<(nh + T - 1) / T, T, 0, stream>>>(menc, nh);
    k_zero<<<(nh + T - 1) / T, T, 0, stream>>>((unsigned*)denom, nh);

    k_edge_alpha<<<(teh + T - 1) / T, T, 0, stream>>>(ei, a_s, a_d, alpha, menc, E, N);
    k_edge_exp<<<(teh + T - 1) / T, T, 0, stream>>>(ei, alpha, menc, denom, E, N);
    k_edge_agg<<<(tew + T - 1) / T, T, 0, stream>>>(ei, xh, alpha, denom, agg, E, N);

    k_ln<<<N, D, 0, stream>>>(agg, att_b + l * D, h, ln_g + l * D, ln_b + l * D, h2);

    k_gemm128<<<rowTiles, 256, 0, stream>>>(h2, wf1, xh, ff1_b + l * D, 1, N);
    k_gemm128<<<rowTiles, 256, 0, stream>>>(xh, wf2, h, ff2_b + l * D, 2, N);
  }

  int nb = 128;
  int per = (N + nb - 1) / nb;
  k_mean1<<<nb, D, 0, stream>>>(h, part, N, per);
  k_mean2<<<1, D, 0, stream>>>(part, (float*)d_out, nb, 1.0f / (float)N);
}
